// DDLIF_44143673868822
// MI455X (gfx1250) — matrix operand order repaired and checked by reading, not compiled, not run on hardware
//
#include <hip/hip_runtime.h>
#include <stdint.h>

typedef float      v4f  __attribute__((ext_vector_type(4)));
typedef float      v8f  __attribute__((ext_vector_type(8)));
typedef _Float16   v16h __attribute__((ext_vector_type(16)));

#define T_STEPS 8
#define BS      8
#define TBDIM   64
#define CDIM    512
#define PIX     1024
#define PAD_W   36

// ---------------------------------------------------------------------------
// Kernel A: LIF scan. One block per (i, c). 256 threads x 4 pixels = 1024 px.
// Writes spikes (u8) and per-(tb, c) spike sums (exact small integers).
// ---------------------------------------------------------------------------
__global__ void lif_scan_kernel(const float* __restrict__ x,
                                uint8_t* __restrict__ sbuf,
                                float* __restrict__ chsum)
{
    const int ic  = blockIdx.x;       // i*CDIM + c
    const int i   = ic >> 9;
    const int c   = ic & 511;
    const int tid = threadIdx.x;
    const int p0  = tid * 4;

    v4f u = {0.f, 0.f, 0.f, 0.f};
    float tsum[T_STEPS];

    for (int t = 0; t < T_STEPS; ++t) {
        size_t off = ((size_t)((t * BS + i) * CDIM + c)) * PIX + p0;
        v4f xv = __builtin_nontemporal_load((const v4f*)(x + off));
        u = 0.5f * u + xv;
        v4f s4;
        s4.x = (u.x > 1.0f) ? 1.0f : 0.0f;
        s4.y = (u.y > 1.0f) ? 1.0f : 0.0f;
        s4.z = (u.z > 1.0f) ? 1.0f : 0.0f;
        s4.w = (u.w > 1.0f) ? 1.0f : 0.0f;
        uint32_t pk = (uint32_t)(s4.x != 0.0f)
                    | ((uint32_t)(s4.y != 0.0f) << 8)
                    | ((uint32_t)(s4.z != 0.0f) << 16)
                    | ((uint32_t)(s4.w != 0.0f) << 24);
        *(uint32_t*)(sbuf + off) = pk;
        u -= s4;
        tsum[t] = s4.x + s4.y + s4.z + s4.w;
    }

    // block-reduce each of the 8 per-t sums (wave32 shuffle + LDS)
    __shared__ float red[T_STEPS][8];
    const int lane = tid & 31, wave = tid >> 5;
    for (int t = 0; t < T_STEPS; ++t) {
        float v = tsum[t];
        for (int o = 16; o > 0; o >>= 1) v += __shfl_down(v, o);
        if (lane == 0) red[t][wave] = v;
    }
    __syncthreads();
    if (tid < T_STEPS) {
        float a = 0.f;
        for (int w = 0; w < 8; ++w) a += red[tid][w];
        chsum[(size_t)(tid * BS + i) * CDIM + c] = a;   // tb = t*BS + i
    }
}

// ---------------------------------------------------------------------------
// Kernel B: sum spikes over channels (partial: 64 channels per block).
// Packed 16-bit field accumulation of u8 data (max 64 per field, exact).
// ---------------------------------------------------------------------------
__global__ void spsum_kernel(const uint8_t* __restrict__ sbuf,
                             float* __restrict__ spart)
{
    const int b = blockIdx.x;          // tb*8 + chunk
    const int tb = b >> 3, chunk = b & 7;
    const int p0 = threadIdx.x * 4;
    uint32_t lo = 0, hi = 0;
    const uint8_t* base = sbuf + ((size_t)tb * CDIM + chunk * 64) * PIX + p0;
    for (int cc = 0; cc < 64; ++cc) {
        uint32_t v = *(const uint32_t*)(base + (size_t)cc * PIX);
        lo += v & 0x00FF00FFu;
        hi += (v >> 8) & 0x00FF00FFu;
    }
    v4f r;
    r.x = (float)(lo & 0xFFFFu);
    r.y = (float)(hi & 0xFFFFu);
    r.z = (float)(lo >> 16);
    r.w = (float)(hi >> 16);
    *(v4f*)(spart + (size_t)b * PIX + p0) = r;
}

// ---------------------------------------------------------------------------
// Kernel C0: total sums -> temporal attention (tiny, 1 block).
// ---------------------------------------------------------------------------
__global__ void temporal_kernel(const float* __restrict__ chsum,
                                const float* __restrict__ wt,
                                float* __restrict__ tem)
{
    __shared__ float part[256];
    __shared__ float tsumS[TBDIM];
    const int tid = threadIdx.x;
    const int tb = tid >> 2, q = tid & 3;
    float a = 0.f;
    for (int c = q * 128; c < q * 128 + 128; ++c) a += chsum[(size_t)tb * CDIM + c];
    part[tid] = a;
    __syncthreads();
    if (q == 0) tsumS[tb] = part[tid] + part[tid + 1] + part[tid + 2] + part[tid + 3];
    __syncthreads();
    if (tid < TBDIM) {
        const int t = tid >> 3, i = tid & 7;
        float acc = 0.f;
        for (int k = 0; k < 5; ++k) {
            int tt = t + k - 2;
            if (tt >= 0 && tt < T_STEPS)
                acc += wt[k] * (tsumS[tt * BS + i] * (1.0f / 524288.0f));
        }
        tem[tid] = acc;
    }
}

// ---------------------------------------------------------------------------
// Kernel C: channel attention (scalar 5-tap) + spatial 5x5 conv as im2col
// GEMM on v_wmma_f32_16x16x32_f16 (K: 50 taps padded to 64; f32 accumulate).
// A-gather is branch-free: tap index clamped to 49; taps >= 50 carry zero
// weight in B so the (finite) clamped value contributes exactly zero.
// One block per tb.
// ---------------------------------------------------------------------------
__global__ void att_kernel(const float* __restrict__ chsum,
                           const float* __restrict__ spart,
                           const float* __restrict__ wch,
                           const float* __restrict__ wsp,
                           float* __restrict__ ch_att,
                           float* __restrict__ sp_att)
{
    __shared__ float chavgP[CDIM + 4];
    __shared__ float spm[PAD_W * PAD_W];
    __shared__ float spx[PAD_W * PAD_W];
    __shared__ float w64[64];
    const int tb = blockIdx.x, tid = threadIdx.x;

    for (int idx = tid; idx < PAD_W * PAD_W; idx += 256) { spm[idx] = 0.f; spx[idx] = 0.f; }
    if (tid < 64) w64[tid] = (tid < 50) ? wsp[tid] : 0.f;
    __syncthreads();

    for (int idx = tid; idx < CDIM + 4; idx += 256)
        chavgP[idx] = (idx >= 2 && idx < CDIM + 2)
                      ? chsum[(size_t)tb * CDIM + idx - 2] * (1.0f / 1024.0f) : 0.f;
    for (int p = tid; p < PIX; p += 256) {
        float s8 = 0.f;
        for (int ch = 0; ch < 8; ++ch) s8 += spart[((size_t)tb * 8 + ch) * PIX + p];
        const int y = p >> 5, xx = p & 31;
        spm[(y + 2) * PAD_W + (xx + 2)] = s8 * (1.0f / 512.0f);   // mean over C
        spx[(y + 2) * PAD_W + (xx + 2)] = (s8 > 0.f) ? 1.f : 0.f; // max over C
    }
    __syncthreads();

    // channel attention
    for (int c = tid; c < CDIM; c += 256) {
        float acc = 0.f;
        for (int k = 0; k < 5; ++k) acc += wch[k] * chavgP[c + k];
        ch_att[(size_t)tb * CDIM + c] = acc;
    }

    // spatial conv via WMMA: D[m][n] = sum_k A[m][k] * B[k][n], B[k][*]=w64[k]
    const int lane = tid & 31, wave = tid >> 5;
    const int nbase = (lane < 16) ? 0 : 8;    // B 32x16 f16: lanes0-15 K=0..7,16..23; lanes16-31 K=8..15,24..31
    v16h b0, b1;
#pragma unroll
    for (int j = 0; j < 16; ++j) {
        b0[j] = (_Float16)w64[nbase + (j < 8 ? j : j + 8)];
        b1[j] = (_Float16)w64[32 + nbase + (j < 8 ? j : j + 8)];
    }
    for (int tile = wave; tile < 64; tile += 8) {      // uniform per wave: EXEC all-ones
        const int m = lane & 15;
        const int pix = tile * 16 + m;
        const int y = pix >> 5, xq = pix & 31;
        v8f acc = {};
        for (int kb = 0; kb < 2; ++kb) {
            v16h a;
#pragma unroll
            for (int j = 0; j < 16; ++j) {
                // A 16x32 f16 layout: lanes0-15 hold K=0..7,16..23; lanes16-31 K=8..15,24..31
                const int klocal = (j < 8 ? j : j + 8) + ((lane < 16) ? 0 : 8);
                const int kg  = kb * 32 + klocal;
                const int kgc = (kg < 50) ? kg : 49;   // clamp: taps >= 50 have zero weight
                const int ch = kgc / 25, r = kgc % 25;
                const int dy = r / 5, dx = r % 5;
                const float val = (ch == 0 ? spm : spx)[(y + dy) * PAD_W + (xq + dx)];
                a[j] = (_Float16)val;
            }
            acc = __builtin_amdgcn_wmma_f32_16x16x32_f16(
                false, a, false, (kb == 0 ? b0 : b1), (short)0, acc, false, false);
        }
        // D layout: VGPR r -> M=r (lanes 0-15), M=r+8 (lanes 16-31); all N equal
        if ((lane & 15) == 0) {
            const int mbase = (lane >> 4) * 8;
#pragma unroll
            for (int r = 0; r < 8; ++r)
                sp_att[(size_t)tb * PIX + tile * 16 + mbase + r] = acc[r];
        }
    }
}

// ---------------------------------------------------------------------------
// Kernel D: output gating, 4 elements/thread, NT stores (134 MB write).
// ---------------------------------------------------------------------------
__global__ void mask_kernel(const uint8_t* __restrict__ sbuf,
                            const float* __restrict__ tem,
                            const float* __restrict__ ch_att,
                            const float* __restrict__ sp_att,
                            float* __restrict__ out)
{
    const size_t idx = ((size_t)blockIdx.x * 256 + threadIdx.x) * 4;
    const int tb  = (int)(idx >> 19);
    const int rem = (int)(idx & 524287);
    const int c   = rem >> 10;
    const int p   = rem & 1023;
    const uint32_t sv = *(const uint32_t*)(sbuf + idx);
    const float tc = tem[tb] * ch_att[(size_t)tb * CDIM + c];
    const v4f sp = *(const v4f*)(sp_att + (size_t)tb * PIX + p);
    v4f o;
    o.x = ((sv & 0xFFu)         && (tc * sp.x >= 0.f)) ? 1.f : 0.f;
    o.y = (((sv >> 8) & 0xFFu)  && (tc * sp.y >= 0.f)) ? 1.f : 0.f;
    o.z = (((sv >> 16) & 0xFFu) && (tc * sp.z >= 0.f)) ? 1.f : 0.f;
    o.w = ((sv >> 24)           && (tc * sp.w >= 0.f)) ? 1.f : 0.f;
    __builtin_nontemporal_store(o, (v4f*)(out + idx));
}

// ---------------------------------------------------------------------------
extern "C" void kernel_launch(void* const* d_in, const int* in_sizes, int n_in,
                              void* d_out, int out_size, void* d_ws, size_t ws_size,
                              hipStream_t stream) {
    const float* x   = (const float*)d_in[0];   // [64,512,32,32]
    const float* wch = (const float*)d_in[1];   // [1,1,5]
    const float* wsp = (const float*)d_in[2];   // [1,2,5,5]
    const float* wt  = (const float*)d_in[3];   // [1,1,5]
    float* out = (float*)d_out;

    uint8_t* ws = (uint8_t*)d_ws;
    const size_t SZ_S     = (size_t)TBDIM * CDIM * PIX;          // 33.5 MB u8 spikes
    uint8_t* sbuf   = ws;
    float* chsum    = (float*)(ws + SZ_S);                       // [64][512]
    float* spart    = (float*)(ws + SZ_S + 131072);              // [64][8][1024]
    float* tem      = (float*)(ws + SZ_S + 131072 + 2097152);    // [64]
    float* ch_att   = (float*)(ws + SZ_S + 131072 + 2097152 + 256);
    float* sp_att   = (float*)(ws + SZ_S + 131072 + 2097152 + 256 + 131072);

    (void)n_in; (void)in_sizes; (void)out_size; (void)ws_size;

    lif_scan_kernel<<<BS * CDIM, 256, 0, stream>>>(x, sbuf, chsum);
    spsum_kernel<<<TBDIM * 8, 256, 0, stream>>>(sbuf, spart);
    temporal_kernel<<<1, 256, 0, stream>>>(chsum, wt, tem);
    att_kernel<<<TBDIM, 256, 0, stream>>>(chsum, spart, wch, wsp, ch_att, sp_att);
    mask_kernel<<<(int)(SZ_S / 1024), 256, 0, stream>>>(sbuf, tem, ch_att, sp_att, out);
}
